// HierarchicalAttentionDecoder_80367428043399
// MI455X (gfx1250) — compile-verified, matrix operand order checked by reading, not run on hardware
//
#include <hip/hip_runtime.h>

// Problem constants (from reference): B=32, S=128, H=512, V=32000, T=64
#define BB 32
#define SS 128
#define HH 512
#define VV 32000
#define TT 64

typedef __bf16 bf16_t;
typedef bf16_t v16bf __attribute__((ext_vector_type(16)));
typedef bf16_t v8bf  __attribute__((ext_vector_type(8)));
typedef float  v8f   __attribute__((ext_vector_type(8)));
typedef unsigned int v4u __attribute__((ext_vector_type(4)));
typedef int v4i __attribute__((ext_vector_type(4)));
typedef int v8i __attribute__((ext_vector_type(8)));

// LDS row stride for the staged B tile: 512 bf16 = 1024B data + 16B TDM pad
// (pad_interval=7 -> pad every 256 DWORDs; pad_amount=3 -> 4 DWORDs)
#define LDS_ROW_BYTES 1040
#define LDS_B_BYTES (128 * LDS_ROW_BYTES)  // 133120

__device__ __forceinline__ float wave_sum(float v) {
  v += __shfl_xor(v, 16, 32);
  v += __shfl_xor(v,  8, 32);
  v += __shfl_xor(v,  4, 32);
  v += __shfl_xor(v,  2, 32);
  v += __shfl_xor(v,  1, 32);
  return v;
}

__device__ __forceinline__ float sigmoidf_(float x) {
  return 1.0f / (1.0f + __expf(-x));
}

// ---------------- init: zero h0 and c0 ----------------
__global__ void k_init(float* __restrict__ h0, float* __restrict__ c0) {
  int i = blockIdx.x * blockDim.x + threadIdx.x;
  if (i < BB * HH) { h0[i] = 0.0f; c0[i] = 0.0f; }
}

// ---------------- W_out fp32 -> bf16 ----------------
__global__ void k_cvt_wout(const float* __restrict__ w, bf16_t* __restrict__ o) {
  for (long i = (long)blockIdx.x * blockDim.x + threadIdx.x; i < (long)VV * HH;
       i += (long)gridDim.x * blockDim.x)
    o[i] = (bf16_t)w[i];
}

// ---------------- transpose [4H, H] -> [H, 4H] ----------------
__global__ void k_transpose(const float* __restrict__ w, float* __restrict__ o) {
  for (long i = (long)blockIdx.x * blockDim.x + threadIdx.x; i < (long)4 * HH * HH;
       i += (long)gridDim.x * blockDim.x) {
    long k = i / (4 * HH);
    long j = i % (4 * HH);
    o[i] = w[j * HH + k];
  }
}

// ---------------- per-step attention + W4 projection -> x[b, :] ----------------
__global__ __launch_bounds__(256) void k_attn(
    const float* __restrict__ wrep,   // [B,S,H]
    const float* __restrict__ srep,   // [B,H]
    const float* __restrict__ W2,     // [H]
    const float* __restrict__ W4,     // [H, 2H]
    const float* __restrict__ h,      // [B,H]
    float* __restrict__ x)            // [B,H]
{
  __shared__ float sh_h[HH];
  __shared__ float sh_sc[SS];
  __shared__ float sh_ctx[HH];
  const int b = blockIdx.x;
  const int tid = threadIdx.x;
  const int lane = tid & 31, wave = tid >> 5;

  for (int k = tid; k < HH; k += 256) sh_h[k] = h[b * HH + k];
  __syncthreads();

  const float* wb = wrep + (size_t)b * SS * HH;

  for (int s = wave; s < SS; s += 8) {
    const float* row = wb + (size_t)s * HH;
    float p = 0.0f;
    for (int k = lane; k < HH; k += 32)
      p += tanhf(row[k] + sh_h[k]) * W2[k];
    p = wave_sum(p);
    if (lane == 0) sh_sc[s] = p;
  }
  __syncthreads();

  if (tid == 0) {
    float m = sh_sc[0];
    for (int s = 1; s < SS; ++s) m = fmaxf(m, sh_sc[s]);
    float sum = 0.0f;
    for (int s = 0; s < SS; ++s) { float e = __expf(sh_sc[s] - m); sh_sc[s] = e; sum += e; }
    float inv = 1.0f / sum;
    for (int s = 0; s < SS; ++s) sh_sc[s] *= inv;
  }
  __syncthreads();

  for (int k = tid; k < HH; k += 256) {
    float acc = 0.0f;
    for (int s = 0; s < SS; ++s) acc += sh_sc[s] * wb[(size_t)s * HH + k];
    sh_ctx[k] = acc;
  }
  __syncthreads();

  const float* sb = srep + (size_t)b * HH;
  for (int j = wave; j < HH; j += 8) {
    const float* wr = W4 + (size_t)j * (2 * HH);
    float p = 0.0f;
    for (int k = lane; k < 2 * HH; k += 32) {
      float cv = (k < HH) ? sb[k] : sh_ctx[k - HH];
      p += cv * wr[k];
    }
    p = wave_sum(p);
    if (lane == 0) x[b * HH + j] = tanhf(p);
  }
}

// ---------------- per-step LSTM gates + cell update ----------------
__global__ __launch_bounds__(256) void k_cell(
    const float* __restrict__ x,
    const float* __restrict__ hprev,
    const float* __restrict__ cin,
    const float* __restrict__ Wit,    // [H, 4H] k-major
    const float* __restrict__ Wht,    // [H, 4H] k-major
    const float* __restrict__ bih,
    const float* __restrict__ bhh,
    float* __restrict__ hnext,
    float* __restrict__ cout,
    bf16_t* __restrict__ hb)
{
  __shared__ float sx[HH];
  __shared__ float sh[HH];
  const int b = blockIdx.x >> 1;
  const int ubase = (blockIdx.x & 1) * 256;
  const int tid = threadIdx.x;

  for (int k = tid; k < HH; k += 256) {
    sx[k] = x[b * HH + k];
    sh[k] = hprev[b * HH + k];
  }
  __syncthreads();

  const int u = ubase + tid;
  float a0 = bih[0 * HH + u] + bhh[0 * HH + u];
  float a1 = bih[1 * HH + u] + bhh[1 * HH + u];
  float a2 = bih[2 * HH + u] + bhh[2 * HH + u];
  float a3 = bih[3 * HH + u] + bhh[3 * HH + u];

  for (int k = 0; k < HH; ++k) {
    const float xv = sx[k], hv = sh[k];
    const float* wi = Wit + (size_t)k * (4 * HH);
    const float* wh = Wht + (size_t)k * (4 * HH);
    a0 += xv * wi[0 * HH + u] + hv * wh[0 * HH + u];
    a1 += xv * wi[1 * HH + u] + hv * wh[1 * HH + u];
    a2 += xv * wi[2 * HH + u] + hv * wh[2 * HH + u];
    a3 += xv * wi[3 * HH + u] + hv * wh[3 * HH + u];
  }

  const float ig = sigmoidf_(a0);
  const float fg = sigmoidf_(a1);
  const float gg = tanhf(a2);
  const float og = sigmoidf_(a3);
  const float cn = fg * cin[b * HH + u] + ig * gg;
  const float hn = og * tanhf(cn);

  hnext[b * HH + u] = hn;
  cout[b * HH + u] = cn;
  hb[b * HH + u] = (bf16_t)hn;
}

// ---------------- batched output projection: bf16 WMMA GEMM + TDM-staged B ----
// A:  [M=2048, K=512] bf16 (row m = t*32 + b), direct global loads
// Bw: [N=32000, K=512] bf16 (W_out rows), staged per-workgroup into LDS via TDM
// Workgroup tile: M128 x N128 (4 M-waves x 2 N-waves; wave tile M32 x N64)
__global__ __launch_bounds__(256) void k_gemm(
    const bf16_t* __restrict__ A,
    const bf16_t* __restrict__ Bw,
    const float* __restrict__ bias,
    float* __restrict__ out)
{
  extern __shared__ char smem[];  // LDS_B_BYTES, row stride LDS_ROW_BYTES

  const int tid = threadIdx.x;
  const int lane = tid & 31, wave = tid >> 5;
  const int wm = wave & 3;        // 4 waves in M
  const int wn = wave >> 2;       // 2 waves in N
  const int ml = lane & 15;
  const int hi = lane >> 4;

  const int mbase = blockIdx.y * 128 + wm * 32;   // 16 groups * 128 = 2048
  const int nwg   = blockIdx.x * 128;             // 250 groups * 128 = 32000

  // ---- stage B tile [128 cols x 512 K] bf16 into LDS with the Tensor Data Mover
  if (tid < 32) {
    const unsigned lds_off = (unsigned)(size_t)smem;  // LDS aperture: low 32 bits = LDS byte offset
    const unsigned long long gaddr =
        (unsigned long long)(const void*)(Bw + (size_t)nwg * HH);

    v4u g0;
    g0[0] = 1u;                                   // count=1, user mode, no gather
    g0[1] = lds_off;                              // lds_addr
    g0[2] = (unsigned)(gaddr & 0xFFFFFFFFu);      // global_addr[31:0]
    g0[3] = (unsigned)((gaddr >> 32) & 0x01FFFFFFu) | (2u << 30);  // addr[56:32] | type=2

    v8i g1;
    // data_size=1 (2B) @[17:16] | pad_enable @[20] | pad_interval=7 @[24:22] | pad_amount=3 @[31:25]
    g1[0] = (int)((1u << 16) | (1u << 20) | (7u << 22) | (3u << 25));
    g1[1] = (int)(((unsigned)HH & 0xFFFFu) << 16);   // tensor_dim0[15:0] @[63:48] = 512
    g1[2] = (int)((128u & 0xFFFFu) << 16);           // tensor_dim1[15:0] @[95:80] = 128
    g1[3] = (int)(((unsigned)HH & 0xFFFFu) << 16);   // tile_dim0 @[127:112] = 512
    g1[4] = (int)128;                                // tile_dim1 @[143:128] = 128 (tile_dim2=0)
    g1[5] = (int)HH;                                 // tensor_dim0_stride[31:0] = 512
    g1[6] = 0;
    g1[7] = 0;

    v4i gz = {0, 0, 0, 0};
#if __clang_major__ >= 23
    v8i gz8 = {0, 0, 0, 0, 0, 0, 0, 0};
    __builtin_amdgcn_tensor_load_to_lds(g0, g1, gz, gz, gz8, 0);
#else
    __builtin_amdgcn_tensor_load_to_lds(g0, g1, gz, gz, 0);
#endif
    __builtin_amdgcn_s_wait_tensorcnt(0);
  }
  __syncthreads();

  const bf16_t* a0row = A + (size_t)(mbase + ml) * HH;        // M-tile 0
  const bf16_t* a1row = A + (size_t)(mbase + 16 + ml) * HH;   // M-tile 1
  const char* bbase = smem + (size_t)(wn * 64 + ml) * LDS_ROW_BYTES;

  v8f acc[2][4];
#pragma unroll
  for (int mt = 0; mt < 2; ++mt)
#pragma unroll
    for (int jt = 0; jt < 4; ++jt) acc[mt][jt] = v8f{};

  for (int kc = 0; kc < HH / 32; ++kc) {
    const int koff = kc * 32 + hi * 8;  // element offset of this lane's first K chunk

    union { v16bf v; v8bf p[2]; } af[2];
    af[0].p[0] = *(const v8bf*)(a0row + koff);
    af[0].p[1] = *(const v8bf*)(a0row + koff + 16);
    af[1].p[0] = *(const v8bf*)(a1row + koff);
    af[1].p[1] = *(const v8bf*)(a1row + koff + 16);

    union { v16bf v; v8bf p[2]; } bfr[4];
#pragma unroll
    for (int jt = 0; jt < 4; ++jt) {
      const char* bp = bbase + (size_t)(jt * 16) * LDS_ROW_BYTES + (size_t)koff * 2;
      bfr[jt].p[0] = *(const v8bf*)(bp);
      bfr[jt].p[1] = *(const v8bf*)(bp + 32);
    }

#pragma unroll
    for (int mt = 0; mt < 2; ++mt)
#pragma unroll
      for (int jt = 0; jt < 4; ++jt)
        acc[mt][jt] = __builtin_amdgcn_wmma_f32_16x16x32_bf16(
            false, af[mt].v, false, bfr[jt].v,
            (short)0, acc[mt][jt], false, false);
  }

  // store with (t,b) -> [B,T,V] permutation and bias
#pragma unroll
  for (int mt = 0; mt < 2; ++mt) {
    const int mtop = mbase + mt * 16 + hi * 8;
#pragma unroll
    for (int jt = 0; jt < 4; ++jt) {
      const int ncol = nwg + wn * 64 + jt * 16 + ml;
      const float bv = bias[ncol];
#pragma unroll
      for (int r = 0; r < 8; ++r) {
        const int m = mtop + r;
        const int b = m & 31;
        const int t = m >> 5;
        out[((size_t)b * TT + t) * VV + ncol] = acc[mt][jt][r] + bv;
      }
    }
  }
}

// ---------------- copy final hidden state into output tail ----------------
__global__ void k_hfin(const float* __restrict__ h, float* __restrict__ o) {
  int i = blockIdx.x * blockDim.x + threadIdx.x;
  if (i < BB * HH) o[i] = h[i];
}

extern "C" void kernel_launch(void* const* d_in, const int* in_sizes, int n_in,
                              void* d_out, int out_size, void* d_ws, size_t ws_size,
                              hipStream_t stream) {
  const float* wrep = (const float*)d_in[0];  // [B,S,H]
  const float* srep = (const float*)d_in[1];  // [B,H]
  const float* W2   = (const float*)d_in[2];  // [H]
  const float* W4   = (const float*)d_in[3];  // [H,2H]
  const float* Wih  = (const float*)d_in[4];  // [4H,H]
  const float* Whh  = (const float*)d_in[5];  // [4H,H]
  const float* bih  = (const float*)d_in[6];  // [4H]
  const float* bhh  = (const float*)d_in[7];  // [4H]
  const float* Wout = (const float*)d_in[8];  // [V,H]
  const float* bout = (const float*)d_in[9];  // [V]
  // d_in[10] = max_len; fixed at TT=64 per reference constants.

  char* ws = (char*)d_ws;
  size_t off = 0;
  float*  h_seq = (float*)(ws + off);  off += (size_t)(TT + 1) * BB * HH * 4;
  float*  cbuf  = (float*)(ws + off);  off += (size_t)2 * BB * HH * 4;
  float*  xbuf  = (float*)(ws + off);  off += (size_t)BB * HH * 4;
  float*  Wit   = (float*)(ws + off);  off += (size_t)4 * HH * HH * 4;
  float*  Wht   = (float*)(ws + off);  off += (size_t)4 * HH * HH * 4;
  bf16_t* hb    = (bf16_t*)(ws + off); off += (size_t)TT * BB * HH * 2;
  bf16_t* wob   = (bf16_t*)(ws + off); off += (size_t)VV * HH * 2;

  k_init<<<(BB * HH + 255) / 256, 256, 0, stream>>>(h_seq, cbuf);
  k_cvt_wout<<<2048, 256, 0, stream>>>(Wout, wob);
  k_transpose<<<1024, 256, 0, stream>>>(Wih, Wit);
  k_transpose<<<1024, 256, 0, stream>>>(Whh, Wht);

  for (int t = 0; t < TT; ++t) {
    const float* hp = h_seq + (size_t)t * BB * HH;
    float*       hn = h_seq + (size_t)(t + 1) * BB * HH;
    const float* ci = cbuf + (size_t)(t & 1) * BB * HH;
    float*       co = cbuf + (size_t)((t + 1) & 1) * BB * HH;
    k_attn<<<BB, 256, 0, stream>>>(wrep, srep, W2, W4, hp, xbuf);
    k_cell<<<2 * BB, 256, 0, stream>>>(xbuf, hp, ci, Wit, Wht, bih, bhh, hn, co,
                                       hb + (size_t)t * BB * HH);
  }

  k_gemm<<<dim3(VV / 128, (TT * BB) / 128), 256, LDS_B_BYTES, stream>>>(
      hb, wob, bout, (float*)d_out);

  k_hfin<<<(BB * HH + 255) / 256, 256, 0, stream>>>(h_seq + (size_t)TT * BB * HH,
                                                    (float*)d_out + (size_t)BB * TT * VV);
}
